// SegMinkUNet_57019985821721
// MI455X (gfx1250) — matrix operand order repaired and checked by reading, not compiled, not run on hardware
//
#include <hip/hip_runtime.h>
#include <hip/hip_bf16.h>

typedef __attribute__((ext_vector_type(16))) _Float16 v16h;
typedef __attribute__((ext_vector_type(8)))  _Float16 v8h;
typedef __attribute__((ext_vector_type(8)))  float    v8f;

#define GD0 48
#define NVX0 (48*48*48)
#define NVX1 (24*24*24)
#define NVX2 (12*12*12)
#define NVX3 (6*6*6)

// ---------------- utility ----------------
__global__ void zero_f32(float* __restrict__ p, long long n) {
    long long i  = (long long)blockIdx.x * blockDim.x + threadIdx.x;
    long long st = (long long)gridDim.x * blockDim.x;
    for (; i < n; i += st) p[i] = 0.f;
}

// ---------------- voxelize ----------------
__global__ void scatter_points(const float* __restrict__ lf, const float* __restrict__ lc,
                               float* __restrict__ sums, float* __restrict__ cnt,
                               int* __restrict__ lin, int N) {
    int p = blockIdx.x * blockDim.x + threadIdx.x;
    if (p >= N) return;
    const float inv = 1.f / 0.05f;
    int i0 = (int)floorf(lc[p*3+0]*inv); i0 = min(max(i0,0),47);
    int i1 = (int)floorf(lc[p*3+1]*inv); i1 = min(max(i1,0),47);
    int i2 = (int)floorf(lc[p*3+2]*inv); i2 = min(max(i2,0),47);
    int l = (i0*48 + i1)*48 + i2;
    lin[p] = l;
    atomicAdd(&sums[l*4+0], lf[p*4+0]);
    atomicAdd(&sums[l*4+1], lf[p*4+1]);
    atomicAdd(&sums[l*4+2], lf[p*4+2]);
    atomicAdd(&sums[l*4+3], lf[p*4+3]);
    atomicAdd(&cnt[l], 1.f);
}

__global__ void vox_norm(const float* __restrict__ sums, const float* __restrict__ cnt,
                         float* __restrict__ vox) {
    int e = blockIdx.x * blockDim.x + threadIdx.x;
    if (e >= NVX0*4) return;
    vox[e] = sums[e] / fmaxf(cnt[e >> 2], 1.f);
}

// ---------------- stem 5x5x5 conv, Cin=4 (VALU; ~2% of FLOPs) ----------------
__global__ void stem_conv5(const float* __restrict__ vox, const float* __restrict__ w,
                           float* __restrict__ out) {
    int idx = blockIdx.x * blockDim.x + threadIdx.x;
    if (idx >= NVX0*32) return;
    int o = idx & 31; int v = idx >> 5;
    int vx = v % 48; int t = v / 48; int vy = t % 48; int vz = t / 48;
    const float* wb = w + (size_t)o*4*125;
    float s = 0.f;
    for (int kz = 0; kz < 5; kz++) {
        int iz = vz + kz - 2; if ((unsigned)iz >= 48u) continue;
        for (int ky = 0; ky < 5; ky++) {
            int iy = vy + ky - 2; if ((unsigned)iy >= 48u) continue;
            for (int kx = 0; kx < 5; kx++) {
                int ix = vx + kx - 2; if ((unsigned)ix >= 48u) continue;
                int tap = (kz*5 + ky)*5 + kx;
                const float* r = vox + (size_t)((iz*48 + iy)*48 + ix)*4;
                s += r[0]*wb[tap] + r[1]*wb[125+tap] + r[2]*wb[250+tap] + r[3]*wb[375+tap];
            }
        }
    }
    out[(size_t)v*32 + o] = s;
}

// ---------------- weight repack: OIDHW f32 -> WMMA B fragments f16 ----------------
// layout: [tap 27][kchunk cin/32][ntile cout/16][lane 32][16 f16]
// lane l: N = l&15, Kbase = 8*(l>>4); element i -> K = chunk*32 + Kbase + (i&7) + 16*(i>>3)
__global__ void repack_w3(const float* __restrict__ w, _Float16* __restrict__ wpk,
                          int cin, int cout) {
    int idx = blockIdx.x * blockDim.x + threadIdx.x;
    int nchunk = cin >> 5, nnt = cout >> 4;
    int total = 27 * nchunk * nnt * 32;
    if (idx >= total) return;
    int lane = idx & 31; int t = idx >> 5;
    int nt = t % nnt; t /= nnt;
    int ch = t % nchunk; int tap = t / nchunk;
    int o = nt*16 + (lane & 15);
    int kbase = ch*32 + ((lane >> 4) << 3);
    _Float16* dst = wpk + (size_t)idx * 16;
#pragma unroll
    for (int i = 0; i < 16; i++)
        dst[i] = (_Float16)w[((size_t)o*cin + (kbase + (i & 7) + ((i >> 3) << 4)))*27 + tap];
}

// ---------------- generic implicit-GEMM 3^3 conv via WMMA ----------------
// in: f16 [in_d^3 + 1 guard row][NCHUNK*32] channels-last (guard row zeroed),
// out: f32 [out_d^3][cout].
// One wave computes a 32-voxel x 32-cout tile: 2 M sub-tiles x 2 N tiles,
// 4 accumulators. A fragments reused across N, B fragments across M ->
// 8 b128 loads per 4 WMMAs. Chunk loop fully unrolled (template); halo taps
// select the guard-row index (v_cndmask).
template<int NCHUNK, int IDIL>
__global__ __launch_bounds__(128)
void conv3d_wmma(const _Float16* __restrict__ in, const _Float16* __restrict__ wpk,
                 float* __restrict__ out,
                 int in_d, int out_d, int cout, int stride, int pad) {
    constexpr unsigned CIN = NCHUNK * 32;
    const int lane  = threadIdx.x & 31;
    const int mtile = blockIdx.x * 4 + (threadIdx.x >> 5);   // 32 output voxels per wave
    const int ntp   = blockIdx.y;                            // pair of 16-wide cout tiles
    const int nnt   = cout >> 4;
    const int nvox  = out_d * out_d * out_d;
    const int half  = lane >> 4;
    const int mrow  = lane & 15;

    const unsigned guard = (unsigned)(in_d * in_d * in_d);   // zeroed guard row

    int vox0 = mtile*32 + mrow;
    int vox1 = vox0 + 16;
    bool mv0 = vox0 < nvox, mv1 = vox1 < nvox;
    int z0=0, y0=0, x0=0, z1=0, y1=0, x1=0;
    if (mv0) { x0 = vox0 % out_d; int t = vox0 / out_d; y0 = t % out_d; z0 = t / out_d; }
    if (mv1) { x1 = vox1 % out_d; int t = vox1 / out_d; y1 = t % out_d; z1 = t / out_d; }

    // per-lane activation row index for a tap; guard row when out of bounds
    auto rowIdx = [&](bool mv, int vz, int vy, int vx,
                      int kz, int ky, int kx) -> unsigned {
        int rz = vz*stride + kz - pad;
        int ry = vy*stride + ky - pad;
        int rx = vx*stride + kx - pad;
        bool ok = mv;
        if (IDIL == 2) { ok = ok && (((rz | ry | rx) & 1) == 0); rz >>= 1; ry >>= 1; rx >>= 1; }
        ok = ok && (unsigned)rz < (unsigned)in_d
                && (unsigned)ry < (unsigned)in_d
                && (unsigned)rx < (unsigned)in_d;
        unsigned o = (unsigned)((rz*in_d + ry)*in_d + rx);
        return ok ? o : guard;                               // v_cndmask
    };

    const size_t bChStride = (size_t)nnt * 512;              // elements per kchunk block
    const _Float16* wtap = wpk + (size_t)ntp*1024 + (size_t)lane*16;

    v8f acc00 = {}, acc01 = {}, acc10 = {}, acc11 = {};

    const _Float16* arow0 = in + (size_t)(rowIdx(mv0, z0, y0, x0, 0, 0, 0) * CIN);
    const _Float16* arow1 = in + (size_t)(rowIdx(mv1, z1, y1, x1, 0, 0, 0) * CIN);
    int kz = 0, ky = 0, kx = 0;
    for (int tap = 0; tap < 27; ++tap) {
        // advance tap coords incrementally; compute + prefetch next rows early
        int nkx = kx + 1, nky = ky, nkz = kz;
        if (nkx == 3) { nkx = 0; nky++; if (nky == 3) { nky = 0; nkz++; } }
        const _Float16* nrow0 = arow0;
        const _Float16* nrow1 = arow1;
        if (tap < 26) {
            nrow0 = in + (size_t)(rowIdx(mv0, z0, y0, x0, nkz, nky, nkx) * CIN);
            nrow1 = in + (size_t)(rowIdx(mv1, z1, y1, x1, nkz, nky, nkx) * CIN);
            __builtin_prefetch(nrow0, 0, 3);
            __builtin_prefetch(nrow1, 0, 3);
        }
#pragma unroll
        for (int ch = 0; ch < NCHUNK; ++ch) {
            const _Float16* ap0 = arow0 + ch*32 + half*8;
            const _Float16* ap1 = arow1 + ch*32 + half*8;
            v8h a0l = *(const v8h*)(ap0);
            v8h a0h = *(const v8h*)(ap0 + 16);
            v8h a1l = *(const v8h*)(ap1);
            v8h a1h = *(const v8h*)(ap1 + 16);
            v16h a0 = __builtin_shufflevector(a0l, a0h,
                      0,1,2,3,4,5,6,7,8,9,10,11,12,13,14,15);
            v16h a1 = __builtin_shufflevector(a1l, a1h,
                      0,1,2,3,4,5,6,7,8,9,10,11,12,13,14,15);
            const _Float16* bp = wtap + (size_t)ch * bChStride;
            v16h b0 = *(const v16h*)(bp);
            v16h b1 = *(const v16h*)(bp + 512);
            acc00 = __builtin_amdgcn_wmma_f32_16x16x32_f16(false, a0, false, b0, (short)0, acc00, false, false);
            acc01 = __builtin_amdgcn_wmma_f32_16x16x32_f16(false, a0, false, b1, (short)0, acc01, false, false);
            acc10 = __builtin_amdgcn_wmma_f32_16x16x32_f16(false, a1, false, b0, (short)0, acc10, false, false);
            acc11 = __builtin_amdgcn_wmma_f32_16x16x32_f16(false, a1, false, b1, (short)0, acc11, false, false);
        }
        arow0 = nrow0; arow1 = nrow1;
        kx = nkx; ky = nky; kz = nkz;
        wtap += (size_t)NCHUNK * bChStride;                  // next tap's weight block
    }

    // C/D layout: lane -> N = lane&15, VGPR r -> M = r + 8*(lane>>4)
    int n = lane & 15;
    int cb = ntp*32 + n;
#pragma unroll
    for (int r = 0; r < 8; r++) {
        int m = r + half*8;
        int va = mtile*32 + m;
        int vb = va + 16;
        if (va < nvox) {
            float* op = out + (size_t)va*cout + cb;
            op[0]  = acc00[r];
            op[16] = acc01[r];
        }
        if (vb < nvox) {
            float* op = out + (size_t)vb*cout + cb;
            op[0]  = acc10[r];
            op[16] = acc11[r];
        }
    }
}

// ---------------- batch-norm stats (sum, sumsq) via LDS float atomics ----------------
__global__ void bn_stats(const float* __restrict__ x, float* __restrict__ stats,
                         long long nelem, int C) {
    __shared__ float ls[512];
    for (int i = threadIdx.x; i < 2*C; i += blockDim.x) ls[i] = 0.f;
    __syncthreads();
    long long i  = (long long)blockIdx.x * blockDim.x + threadIdx.x;
    long long st = (long long)gridDim.x * blockDim.x;
    for (; i < nelem; i += st) {
        int c = (int)(i % C);
        float v = x[i];
        atomicAdd(&ls[c], v);
        atomicAdd(&ls[C + c], v*v);
    }
    __syncthreads();
    for (int i2 = threadIdx.x; i2 < 2*C; i2 += blockDim.x) atomicAdd(&stats[i2], ls[i2]);
}

// ---------------- fused BN + ReLU + f16 convert + concat placement ----------------
__global__ void bn_apply(const float* __restrict__ x, const float* __restrict__ stats,
                         const float* __restrict__ g, const float* __restrict__ b,
                         _Float16* __restrict__ dst, long long nvox, int C,
                         int dstStride, int dstOff, float invN) {
    long long i  = (long long)blockIdx.x * blockDim.x + threadIdx.x;
    long long st = (long long)gridDim.x * blockDim.x;
    long long n  = nvox * C;
    for (; i < n; i += st) {
        int c = (int)(i % C);
        long long v = i / C;
        float m   = stats[c] * invN;
        float var = fmaxf(stats[C + c] * invN - m*m, 0.f);
        float xn  = (x[i] - m) * rsqrtf(var + 1e-5f);
        float y   = fmaxf(xn * g[c] + b[c], 0.f);
        dst[v * dstStride + dstOff + c] = (_Float16)y;
    }
}

// ---------------- skip copy into concat buffer ----------------
__global__ void copy_skip(const _Float16* __restrict__ src, int srcStride,
                          _Float16* __restrict__ dst, int dstStride, int dstOff,
                          long long nvox, int C) {
    long long i  = (long long)blockIdx.x * blockDim.x + threadIdx.x;
    long long st = (long long)gridDim.x * blockDim.x;
    long long n  = nvox * C;
    for (; i < n; i += st) {
        int c = (int)(i % C);
        long long v = i / C;
        dst[v * dstStride + dstOff + c] = src[v * srcStride + c];
    }
}

// ---------------- per-point gather + L2 normalize (one wave32 per point) ----------------
__global__ void gather_norm(const _Float16* __restrict__ feat, const int* __restrict__ lin,
                            float* __restrict__ out, int N) {
    int tid  = blockIdx.x * blockDim.x + threadIdx.x;
    int wid  = tid >> 5;
    int lane = tid & 31;
    if (wid >= N) return;
    const _Float16* f = feat + (size_t)lin[wid] * 96;
    float a = (float)f[lane], b = (float)f[lane + 32], c = (float)f[lane + 64];
    float ss = a*a + b*b + c*c;
    for (int o = 16; o > 0; o >>= 1) ss += __shfl_xor(ss, o, 32);
    float inv = rsqrtf(ss);
    float* op = out + (size_t)wid * 96;
    op[lane] = a * inv; op[lane + 32] = b * inv; op[lane + 64] = c * inv;
}

// =====================================================================
extern "C" void kernel_launch(void* const* d_in, const int* in_sizes, int n_in,
                              void* d_out, int out_size, void* d_ws, size_t ws_size,
                              hipStream_t stream) {
    const int N = in_sizes[0] / 4;

    const float* lidar_f = (const float*)d_in[0];
    const float* lidar_c = (const float*)d_in[1];
    const float* stem_w0 = (const float*)d_in[2];
    const float* stem_g0 = (const float*)d_in[3];
    const float* stem_b0 = (const float*)d_in[4];
    const float* stem_w1 = (const float*)d_in[5];
    const float* stem_g1 = (const float*)d_in[6];
    const float* stem_b1 = (const float*)d_in[7];
    const float* dn[3][6]; // w0 g0 b0 w1 g1 b1
    for (int i = 0; i < 3; i++)
        for (int j = 0; j < 6; j++) dn[i][j] = (const float*)d_in[8 + i*6 + j];
    const float* up[3][6]; // wt gt bt w1 g1 b1
    for (int i = 0; i < 3; i++)
        for (int j = 0; j < 6; j++) up[i][j] = (const float*)d_in[26 + i*6 + j];

    // ---- workspace carve (each f16 act buffer gets a 512B guard row slot) ----
    char* ws = (char*)d_ws;
    size_t off = 0;
    auto take = [&](size_t bytes) { size_t o = off; off = (off + bytes + 255) & ~(size_t)255; return o; };
    _Float16* WPK  = (_Float16*)(ws + take((size_t)27*256*128*2)); // max packed weights
    float*    STATS= (float*)   (ws + take(2048));
    float*    SUMS = (float*)   (ws + take((size_t)NVX0*4*4));
    float*    CNT  = (float*)   (ws + take((size_t)NVX0*4));
    float*    VOX  = (float*)   (ws + take((size_t)NVX0*4*4));
    float*    RAW  = (float*)   (ws + take((size_t)NVX0*96*4));    // largest f32 conv output
    _Float16* ACTA = (_Float16*)(ws + take((size_t)NVX0*128*2 + 512));  // +guard
    _Float16* ACTB = (_Float16*)(ws + take((size_t)NVX0*96*2  + 512));
    _Float16* SK1  = (_Float16*)(ws + take((size_t)NVX0*32*2  + 512));
    _Float16* SK2  = (_Float16*)(ws + take((size_t)NVX1*64*2  + 512));
    _Float16* SK3  = (_Float16*)(ws + take((size_t)NVX2*128*2 + 512));
    int*      LIN  = (int*)     (ws + take((size_t)N*4));

    auto run_bn = [&](const float* raw, const float* g, const float* b,
                      int nvox, int C, _Float16* dst, int dstStride, int dstOff) {
        zero_f32<<<1, 256, 0, stream>>>(STATS, 2*C);
        long long ne = (long long)nvox * C;
        int gb = (int)((ne + 255) / 256); if (gb > 2048) gb = 2048;
        bn_stats<<<gb, 256, 0, stream>>>(raw, STATS, ne, C);
        int ga = (int)((ne + 255) / 256); if (ga > 4096) ga = 4096;
        bn_apply<<<ga, 256, 0, stream>>>(raw, STATS, g, b, dst, (long long)nvox, C,
                                         dstStride, dstOff, 1.f / (float)nvox);
    };

    auto run_conv = [&](const _Float16* inp, const float* w, const float* g, const float* b,
                        int in_d, int out_d, int cin, int cout,
                        int stride, int idil, int pad,
                        _Float16* dst, int dstStride, int dstOff) {
        int nchunk = cin / 32, nnt = cout / 16;
        int tot = 27 * nchunk * nnt * 32;
        repack_w3<<<(tot + 255) / 256, 256, 0, stream>>>(w, WPK, cin, cout);
        // zero the input buffer's guard row (row index in_d^3, cin f16 = cin/2 f32)
        long long nvi = (long long)in_d * in_d * in_d;
        zero_f32<<<1, 64, 0, stream>>>((float*)((_Float16*)inp + nvi * cin), cin / 2);
        int nvox = out_d * out_d * out_d;
        int mt32 = (nvox + 31) / 32;
        dim3 cg((mt32 + 3) / 4, cout / 32);
#define CONV_CASE(NC, ID) conv3d_wmma<NC, ID><<<cg, 128, 0, stream>>>( \
            inp, WPK, RAW, in_d, out_d, cout, stride, pad)
        if (idil == 1) {
            switch (nchunk) {
                case 1: CONV_CASE(1, 1); break;
                case 2: CONV_CASE(2, 1); break;
                case 3: CONV_CASE(3, 1); break;
                case 4: CONV_CASE(4, 1); break;
                case 5: CONV_CASE(5, 1); break;
                default: CONV_CASE(8, 1); break;
            }
        } else {
            switch (nchunk) {
                case 3: CONV_CASE(3, 2); break;
                case 4: CONV_CASE(4, 2); break;
                default: CONV_CASE(8, 2); break;
            }
        }
#undef CONV_CASE
        run_bn(RAW, g, b, nvox, cout, dst, dstStride, dstOff);
    };

    auto run_copy = [&](const _Float16* src, int srcStride, _Float16* dst,
                        int dstStride, int dstOff, int nvox, int C) {
        long long ne = (long long)nvox * C;
        int gb = (int)((ne + 255) / 256); if (gb > 4096) gb = 4096;
        copy_skip<<<gb, 256, 0, stream>>>(src, srcStride, dst, dstStride, dstOff,
                                          (long long)nvox, C);
    };

    // ---- voxelize (scatter mean) ----
    zero_f32<<<512, 256, 0, stream>>>(SUMS, (long long)NVX0*4);
    zero_f32<<<256, 256, 0, stream>>>(CNT,  (long long)NVX0);
    scatter_points<<<(N + 255) / 256, 256, 0, stream>>>(lidar_f, lidar_c, SUMS, CNT, LIN, N);
    vox_norm<<<(NVX0*4 + 255) / 256, 256, 0, stream>>>(SUMS, CNT, VOX);

    // ---- stem ----
    stem_conv5<<<(NVX0*32 + 255) / 256, 256, 0, stream>>>(VOX, stem_w0, RAW);
    run_bn(RAW, stem_g0, stem_b0, NVX0, 32, ACTA, 32, 0);
    run_conv(ACTA, stem_w1, stem_g1, stem_b1, 48, 48, 32, 32, 1, 1, 1, SK1, 32, 0); // v1

    // ---- down path ----
    run_conv(SK1,  dn[0][0], dn[0][1], dn[0][2], 48, 24,  32,  64, 2, 1, 0, ACTA,  64, 0);
    run_conv(ACTA, dn[0][3], dn[0][4], dn[0][5], 24, 24,  64,  64, 1, 1, 1, SK2,   64, 0);
    run_conv(SK2,  dn[1][0], dn[1][1], dn[1][2], 24, 12,  64, 128, 2, 1, 0, ACTA, 128, 0);
    run_conv(ACTA, dn[1][3], dn[1][4], dn[1][5], 12, 12, 128, 128, 1, 1, 1, SK3,  128, 0);
    run_conv(SK3,  dn[2][0], dn[2][1], dn[2][2], 12,  6, 128, 256, 2, 1, 0, ACTA, 256, 0);
    run_conv(ACTA, dn[2][3], dn[2][4], dn[2][5],  6,  6, 256, 256, 1, 1, 1, ACTB, 256, 0);

    // ---- up path (transposed conv = input-dilated conv, idil=2) ----
    run_conv(ACTB, up[0][0], up[0][1], up[0][2],  6, 12, 256, 128, 1, 2, 1, ACTA, 256, 0);
    run_copy(SK3, 128, ACTA, 256, 128, NVX2, 128);
    run_conv(ACTA, up[0][3], up[0][4], up[0][5], 12, 12, 256, 128, 1, 1, 1, ACTB, 128, 0);

    run_conv(ACTB, up[1][0], up[1][1], up[1][2], 12, 24, 128,  96, 1, 2, 1, ACTA, 160, 0);
    run_copy(SK2, 64, ACTA, 160, 96, NVX1, 64);
    run_conv(ACTA, up[1][3], up[1][4], up[1][5], 24, 24, 160,  96, 1, 1, 1, ACTB, 96, 0);

    run_conv(ACTB, up[2][0], up[2][1], up[2][2], 24, 48,  96,  96, 1, 2, 1, ACTA, 128, 0);
    run_copy(SK1, 32, ACTA, 128, 96, NVX0, 32);
    run_conv(ACTA, up[2][3], up[2][4], up[2][5], 48, 48, 128,  96, 1, 1, 1, ACTB, 96, 0);

    // ---- gather + normalize ----
    int gblocks = ((N * 32) + 255) / 256;
    gather_norm<<<gblocks, 256, 0, stream>>>(ACTB, LIN, (float*)d_out, N);
}
